// Attention_Decoder_5978594476035
// MI455X (gfx1250) — matrix operand order repaired and checked by reading, not compiled, not run on hardware
//
#include <hip/hip_runtime.h>
#include <hip/hip_bf16.h>
#include <math.h>

// ---------------------------------------------------------------------------
// CDNA5 / gfx1250, wave32. GEMVs go through v_wmma_f32_16x16x32_f16 with an
// fp32->2xfp16 split (3 WMMA products) for f32-class accuracy. Workload is
// HBM-bound (~140MB fp32 weights, ~72 MFLOP): floor ~6us at 23.3 TB/s.
// Weights (~140MB) fit in the 192MB global L2, so default RT temporal hints
// keep them resident across graph replays. K templated + unroll-4 so each
// wave keeps ~8KB of global_load_b128 in flight to cover HBM latency.
// ---------------------------------------------------------------------------

typedef __attribute__((ext_vector_type(16))) _Float16 v16h;
typedef __attribute__((ext_vector_type(8)))  float    v8f;

__device__ __forceinline__ float wave_sum32(float v) {
    #pragma unroll
    for (int off = 16; off; off >>= 1) v += __shfl_xor(v, off, 32);
    return v;
}
__device__ __forceinline__ float wave_max32(float v) {
    #pragma unroll
    for (int off = 16; off; off >>= 1) v = fmaxf(v, __shfl_xor(v, off, 32));
    return v;
}

// ---------------------------------------------------------------------------
// WMMA GEMV: y[row0 .. row0+15] = W[row, :] . x + bias  (K multiple of 32)
// One wave per 16-row tile. B = x replicated across all 16 columns, so every
// result column equals the GEMV; lanes 0 / 16 extract rows 0-7 / 8-15.
// ---------------------------------------------------------------------------
template<int K>
__global__ __launch_bounds__(256) void k_gemv_wmma(
        const float* __restrict__ W, const float* __restrict__ x,
        const float* __restrict__ bias, float* __restrict__ y,
        int nrows)
{
    const int ntiles = (nrows + 15) >> 4;
    const int tile = blockIdx.x * 8 + (threadIdx.x >> 5);
    if (tile >= ntiles) return;                 // uniform per-wave
    const int lane = threadIdx.x & 31;
    const int hl   = lane >> 4;                 // half: 0 or 1
    const int m    = lane & 15;
    const int row0 = tile << 4;
    const int nvalid = min(16, nrows - row0);
    const int rload = (m < nvalid) ? (row0 + m) : row0;   // clamp OOB row loads
    const float* __restrict__ wrow = W + (size_t)rload * (size_t)K;

    v8f acc = {0.f, 0.f, 0.f, 0.f, 0.f, 0.f, 0.f, 0.f};

    #pragma unroll 4
    for (int k0 = 0; k0 < K; k0 += 32) {
        // ---- A tile 16x32 fp16 (documented layout), split hi/lo
        const int ka = k0 + hl * 8;
        const float4* pa0 = (const float4*)(wrow + ka);        // K = ka..ka+7
        const float4* pa1 = (const float4*)(wrow + ka + 16);   // K = ka+16..ka+23
        float wf[16];
        float4 t;
        t = pa0[0]; wf[0]=t.x; wf[1]=t.y; wf[2]=t.z; wf[3]=t.w;
        t = pa0[1]; wf[4]=t.x; wf[5]=t.y; wf[6]=t.z; wf[7]=t.w;
        t = pa1[0]; wf[8]=t.x; wf[9]=t.y; wf[10]=t.z; wf[11]=t.w;
        t = pa1[1]; wf[12]=t.x; wf[13]=t.y; wf[14]=t.z; wf[15]=t.w;
        v16h ahi, alo;
        #pragma unroll
        for (int j = 0; j < 16; ++j) {
            _Float16 h = (_Float16)wf[j];
            ahi[j] = h;
            alo[j] = (_Float16)(wf[j] - (float)h);
        }
        // ---- B tile 32x16 fp16, x replicated over columns, split hi/lo
        const int kb = k0 + hl * 8;
        const float4* pb = (const float4*)(x + kb);
        float xf[16];
        t = pb[0]; xf[0]=t.x; xf[1]=t.y; xf[2]=t.z; xf[3]=t.w;
        t = pb[1]; xf[4]=t.x; xf[5]=t.y; xf[6]=t.z; xf[7]=t.w;
        t = pb[4]; xf[8]=t.x; xf[9]=t.y; xf[10]=t.z; xf[11]=t.w;
        t = pb[5]; xf[12]=t.x; xf[13]=t.y; xf[14]=t.z; xf[15]=t.w;
        v16h bhi, blo;
        #pragma unroll
        for (int j = 0; j < 16; ++j) {
            _Float16 h = (_Float16)xf[j];
            bhi[j] = h;
            blo[j] = (_Float16)(xf[j] - (float)h);
        }
        // ---- 3-product split accumulation (f32 accumulator)
        acc = __builtin_amdgcn_wmma_f32_16x16x32_f16(false, alo, false, bhi, (short)0, acc, false, false);
        acc = __builtin_amdgcn_wmma_f32_16x16x32_f16(false, ahi, false, blo, (short)0, acc, false, false);
        acc = __builtin_amdgcn_wmma_f32_16x16x32_f16(false, ahi, false, bhi, (short)0, acc, false, false);
    }

    if (m == 0) {       // lane 0 -> rows 0..7 ; lane 16 -> rows 8..15
        #pragma unroll
        for (int v = 0; v < 8; ++v) {
            int rr = hl * 8 + v;
            if (rr < nvalid) {
                int r = row0 + rr;
                y[r] = acc[v] + bias[r];
            }
        }
    }
}

// ---------------------------------------------------------------------------
// Embedding gather: embv[0..511] = emb[token, :]
// ---------------------------------------------------------------------------
__global__ void k_embed(const float* __restrict__ emb, const int* __restrict__ x,
                        float* __restrict__ embv)
{
    int tok = x[0];
    embv[threadIdx.x] = emb[(size_t)tok * 512 + threadIdx.x];
}

// ---------------------------------------------------------------------------
// Conv stack: pre = [in0 (D-1 ch), embv] -> conv(32ch)+BN+ReLU ->
// conv(Dch)+residual+BN+ReLU -> conv(1ch)+BN+ReLU -> outvec[512]
// One workgroup of 512 threads (16 waves); BN stats via wave32 reductions.
// ---------------------------------------------------------------------------
template<int D>
__global__ __launch_bounds__(512) void k_convstack(
        const float* __restrict__ in0, const float* __restrict__ embv,
        const float* __restrict__ w1, const float* __restrict__ b1,  // (32,D,3)
        const float* __restrict__ w2, const float* __restrict__ b2,  // (D,32,3)
        const float* __restrict__ w3, const float* __restrict__ b3,  // (1,D,3)
        const float* __restrict__ g1, const float* __restrict__ bb1,
        const float* __restrict__ g2, const float* __restrict__ bb2,
        const float* __restrict__ g3, const float* __restrict__ bb3,
        float* __restrict__ outvec)
{
    __shared__ float s_pre[D][512];
    __shared__ float s_c[32][512];
    __shared__ float s_p2[D][512];
    __shared__ float s_mean[32], s_rstd[32];

    const int tid  = threadIdx.x;
    const int wave = tid >> 5;
    const int lane = tid & 31;
    const float inv512 = 1.0f / 512.0f;

    #pragma unroll
    for (int i = 0; i < D - 1; ++i) s_pre[i][tid] = in0[i * 512 + tid];
    s_pre[D - 1][tid] = embv[tid];
    __syncthreads();

    // conv1 (32 out channels)
    for (int idx = tid; idx < 32 * 512; idx += 512) {
        int o = idx >> 9, l = idx & 511;
        float acc = b1[o];
        #pragma unroll
        for (int i = 0; i < D; ++i) {
            const float* wr = &w1[(o * D + i) * 3];
            float xm = (l > 0)   ? s_pre[i][l - 1] : 0.f;
            float x0 = s_pre[i][l];
            float xp = (l < 511) ? s_pre[i][l + 1] : 0.f;
            acc += wr[0] * xm + wr[1] * x0 + wr[2] * xp;
        }
        s_c[o][l] = acc;
    }
    __syncthreads();

    // BN1 stats: each wave handles 2 channels
    for (int o = wave; o < 32; o += 16) {
        float s = 0.f, sq = 0.f;
        #pragma unroll
        for (int j = 0; j < 16; ++j) { float v = s_c[o][lane + 32 * j]; s += v; sq += v * v; }
        s = wave_sum32(s); sq = wave_sum32(sq);
        if (lane == 0) {
            float mu = s * inv512;
            float var = sq * inv512 - mu * mu;
            s_mean[o] = mu; s_rstd[o] = rsqrtf(var + 1e-5f);
        }
    }
    __syncthreads();
    for (int idx = tid; idx < 32 * 512; idx += 512) {
        int o = idx >> 9, l = idx & 511;
        float v = (s_c[o][l] - s_mean[o]) * s_rstd[o] * g1[o] + bb1[o];
        s_c[o][l] = fmaxf(v, 0.f);
    }
    __syncthreads();

    // conv2 + residual (D out channels)
    {
        int l = tid;
        #pragma unroll
        for (int o = 0; o < D; ++o) {
            float acc = b2[o] + s_pre[o][l];
            for (int i = 0; i < 32; ++i) {
                const float* wr = &w2[(o * 32 + i) * 3];
                float xm = (l > 0)   ? s_c[i][l - 1] : 0.f;
                float x0 = s_c[i][l];
                float xp = (l < 511) ? s_c[i][l + 1] : 0.f;
                acc += wr[0] * xm + wr[1] * x0 + wr[2] * xp;
            }
            s_p2[o][l] = acc;
        }
    }
    __syncthreads();
    if (wave < D) {
        int o = wave;
        float s = 0.f, sq = 0.f;
        #pragma unroll
        for (int j = 0; j < 16; ++j) { float v = s_p2[o][lane + 32 * j]; s += v; sq += v * v; }
        s = wave_sum32(s); sq = wave_sum32(sq);
        if (lane == 0) {
            float mu = s * inv512;
            float var = sq * inv512 - mu * mu;
            s_mean[o] = mu; s_rstd[o] = rsqrtf(var + 1e-5f);
        }
    }
    __syncthreads();
    #pragma unroll
    for (int o = 0; o < D; ++o) {
        float v = (s_p2[o][tid] - s_mean[o]) * s_rstd[o] * g2[o] + bb2[o];
        s_p2[o][tid] = fmaxf(v, 0.f);
    }
    __syncthreads();

    // conv3 (1 out channel) into s_c[0]
    {
        int l = tid;
        float acc = b3[0];
        #pragma unroll
        for (int i = 0; i < D; ++i) {
            const float* wr = &w3[i * 3];
            float xm = (l > 0)   ? s_p2[i][l - 1] : 0.f;
            float x0 = s_p2[i][l];
            float xp = (l < 511) ? s_p2[i][l + 1] : 0.f;
            acc += wr[0] * xm + wr[1] * x0 + wr[2] * xp;
        }
        s_c[0][l] = acc;
    }
    __syncthreads();
    if (wave == 0) {
        float s = 0.f, sq = 0.f;
        #pragma unroll
        for (int j = 0; j < 16; ++j) { float v = s_c[0][lane + 32 * j]; s += v; sq += v * v; }
        s = wave_sum32(s); sq = wave_sum32(sq);
        if (lane == 0) {
            float mu = s * inv512;
            float var = sq * inv512 - mu * mu;
            s_mean[0] = mu; s_rstd[0] = rsqrtf(var + 1e-5f);
        }
    }
    __syncthreads();
    outvec[tid] = fmaxf((s_c[0][tid] - s_mean[0]) * s_rstd[0] * g3[0] + bb3[0], 0.f);
}

// ---------------------------------------------------------------------------
// Softmax over 2048 scores (in place), one workgroup of 256
// ---------------------------------------------------------------------------
__global__ __launch_bounds__(256) void k_softmax2048(float* __restrict__ s)
{
    __shared__ float red[8];
    __shared__ float gmax_s, gsum_s;
    const int tid = threadIdx.x, wave = tid >> 5, lane = tid & 31;

    float lmax = -3.0e38f;
    for (int i = tid; i < 2048; i += 256) lmax = fmaxf(lmax, s[i]);
    lmax = wave_max32(lmax);
    if (lane == 0) red[wave] = lmax;
    __syncthreads();
    if (tid == 0) {
        float g = red[0];
        #pragma unroll
        for (int w = 1; w < 8; ++w) g = fmaxf(g, red[w]);
        gmax_s = g;
    }
    __syncthreads();
    const float gmax = gmax_s;

    float lsum = 0.f;
    for (int i = tid; i < 2048; i += 256) lsum += expf(s[i] - gmax);
    lsum = wave_sum32(lsum);
    if (lane == 0) red[wave] = lsum;
    __syncthreads();
    if (tid == 0) {
        float g = 0.f;
        #pragma unroll
        for (int w = 0; w < 8; ++w) g += red[w];
        gsum_s = g;
    }
    __syncthreads();
    const float inv = 1.0f / gsum_s;
    for (int i = tid; i < 2048; i += 256) s[i] = expf(s[i] - gmax) * inv;
}

// ---------------------------------------------------------------------------
// app[j] = sum_i aw[i] * enc[i][j]   (enc: 2048 x 1024 row-major)
// ---------------------------------------------------------------------------
__global__ __launch_bounds__(256) void k_apply_att(
        const float* __restrict__ aw, const float* __restrict__ enc,
        float* __restrict__ app)
{
    const int j = blockIdx.x * 256 + threadIdx.x;    // 0..1023
    float acc = 0.f;
    for (int i = 0; i < 2048; ++i) acc = fmaf(aw[i], enc[(size_t)i * 1024 + j], acc);
    app[j] = acc;
}

// ---------------------------------------------------------------------------
// GRU gate apply (PyTorch order r,z,n), one layer, 512 threads
// ---------------------------------------------------------------------------
__global__ __launch_bounds__(512) void k_gru_gate(
        const float* __restrict__ gi, const float* __restrict__ gh,
        const float* __restrict__ hprev, float* __restrict__ hnew)
{
    const int t = threadIdx.x;
    float r = 1.f / (1.f + expf(-(gi[t]        + gh[t])));
    float z = 1.f / (1.f + expf(-(gi[512 + t]  + gh[512 + t])));
    float n = tanhf(gi[1024 + t] + r * gh[1024 + t]);
    hnew[t] = (1.f - z) * n + z * hprev[t];
}

// ---------------------------------------------------------------------------
// log-softmax over V=50257: reduce (max, log-sum-exp) with one 1024-thread WG
// ---------------------------------------------------------------------------
__global__ __launch_bounds__(1024) void k_lsm_reduce(
        const float* __restrict__ logits, float* __restrict__ red2, int V)
{
    __shared__ float sm[32];
    __shared__ float gmax_s, gsum_s;
    const int tid = threadIdx.x, wave = tid >> 5, lane = tid & 31;

    float lmax = -3.0e38f;
    for (int i = tid; i < V; i += 1024) lmax = fmaxf(lmax, logits[i]);
    lmax = wave_max32(lmax);
    if (lane == 0) sm[wave] = lmax;
    __syncthreads();
    if (tid < 32) {
        float v = sm[tid];
        v = wave_max32(v);
        if (tid == 0) gmax_s = v;
    }
    __syncthreads();
    const float gmax = gmax_s;

    float lsum = 0.f;
    for (int i = tid; i < V; i += 1024) lsum += expf(logits[i] - gmax);
    lsum = wave_sum32(lsum);
    if (lane == 0) sm[wave] = lsum;
    __syncthreads();
    if (tid < 32) {
        float v = sm[tid];
        v = wave_sum32(v);
        if (tid == 0) gsum_s = v;
    }
    __syncthreads();
    if (tid == 0) { red2[0] = gmax; red2[1] = logf(gsum_s); }
}

__global__ __launch_bounds__(256) void k_lsm_apply(
        const float* __restrict__ logits, const float* __restrict__ red2,
        float* __restrict__ out, int V)
{
    const int i = blockIdx.x * 256 + threadIdx.x;
    if (i < V) out[i] = logits[i] - red2[0] - red2[1];
}

// ---------------------------------------------------------------------------
// Host-side orchestration (graph-capturable: only kernel launches on stream)
// ---------------------------------------------------------------------------
extern "C" void kernel_launch(void* const* d_in, const int* in_sizes, int n_in,
                              void* d_out, int out_size, void* d_ws, size_t ws_size,
                              hipStream_t stream) {
    (void)in_sizes; (void)n_in; (void)out_size; (void)ws_size;

    const float* emb     = (const float*)d_in[0];
    const float* conv1_w = (const float*)d_in[1];
    const float* conv1_b = (const float*)d_in[2];
    const float* conv2_w = (const float*)d_in[3];
    const float* conv2_b = (const float*)d_in[4];
    const float* conv3_w = (const float*)d_in[5];
    const float* conv3_b = (const float*)d_in[6];
    const float* conv4_w = (const float*)d_in[7];
    const float* conv4_b = (const float*)d_in[8];
    const float* conv5_w = (const float*)d_in[9];
    const float* conv5_b = (const float*)d_in[10];
    const float* conv6_w = (const float*)d_in[11];
    const float* conv6_b = (const float*)d_in[12];
    const float* bn1_g = (const float*)d_in[13];
    const float* bn1_b = (const float*)d_in[14];
    const float* bn2_g = (const float*)d_in[15];
    const float* bn2_b = (const float*)d_in[16];
    const float* bn3_g = (const float*)d_in[17];
    const float* bn3_b = (const float*)d_in[18];
    const float* bn4_g = (const float*)d_in[19];
    const float* bn4_b = (const float*)d_in[20];
    const float* bn5_g = (const float*)d_in[21];
    const float* bn5_b = (const float*)d_in[22];
    const float* bn6_g = (const float*)d_in[23];
    const float* bn6_b = (const float*)d_in[24];
    const float* att_w = (const float*)d_in[25];
    const float* att_b = (const float*)d_in[26];
    const float* gru_wih = (const float*)d_in[27];
    const float* gru_whh = (const float*)d_in[28];
    const float* gru_bih = (const float*)d_in[29];
    const float* gru_bhh = (const float*)d_in[30];
    const float* lo_w  = (const float*)d_in[31];
    const float* lo_b  = (const float*)d_in[32];
    const float* h_st  = (const float*)d_in[33];
    const float* enc   = (const float*)d_in[34];
    const int*   x     = (const int*)d_in[35];

    const int V = 50257;
    float* ws = (float*)d_ws;
    float* embv   = ws;                 // 512
    float* prevec = ws + 512;           // 512
    float* aw     = ws + 1024;          // 2048 (scores -> softmax weights)
    float* app    = ws + 3072;          // 1024
    float* comvec = ws + 4096;          // 512
    float* gi     = ws + 5120;          // 1536
    float* gh     = ws + 6656;          // 1536
    float* logits = ws + 8192;          // 50257 (+pad)
    float* red2   = ws + 8192 + 50304;  // 2

    float* outv = (float*)d_out;        // [0,V): log-probs ; [V, V+2048): h_new

    // Stage 1: embedding + conv stack 1 -> prevec
    k_embed<<<1, 512, 0, stream>>>(emb, x, embv);
    k_convstack<5><<<1, 512, 0, stream>>>(h_st, embv,
        conv1_w, conv1_b, conv2_w, conv2_b, conv3_w, conv3_b,
        bn1_g, bn1_b, bn2_g, bn2_b, bn3_g, bn3_b, prevec);

    // Stage 2: attention scores (WMMA GEMV, 2048x512), softmax, apply
    k_gemv_wmma<512><<<16, 256, 0, stream>>>(att_w, prevec, att_b, aw, 2048);
    k_softmax2048<<<1, 256, 0, stream>>>(aw);
    k_apply_att<<<4, 256, 0, stream>>>(aw, enc, app);

    // Stage 3: conv stack 2 -> comvec
    k_convstack<3><<<1, 512, 0, stream>>>(app, embv,
        conv4_w, conv4_b, conv5_w, conv5_b, conv6_w, conv6_b,
        bn4_g, bn4_b, bn5_g, bn5_b, bn6_g, bn6_b, comvec);

    // Stage 4: 4-layer GRU (two 1536x512 WMMA GEMVs + gate apply per layer)
    const float* xt = comvec;
    for (int l = 0; l < 4; ++l) {
        const size_t woff = (size_t)l * 1536 * 512;
        k_gemv_wmma<512><<<12, 256, 0, stream>>>(gru_wih + woff, xt,
                                                 gru_bih + (size_t)l * 1536, gi, 1536);
        k_gemv_wmma<512><<<12, 256, 0, stream>>>(gru_whh + woff, h_st + (size_t)l * 512,
                                                 gru_bhh + (size_t)l * 1536, gh, 1536);
        k_gru_gate<<<1, 512, 0, stream>>>(gi, gh, h_st + (size_t)l * 512,
                                          outv + V + (size_t)l * 512);
        xt = outv + V + (size_t)l * 512;
    }

    // Stage 5: vocab projection (50257x512, bandwidth-dominant) + log-softmax
    k_gemv_wmma<512><<<393, 256, 0, stream>>>(lo_w, xt, lo_b, logits, V);
    k_lsm_reduce<<<1, 1024, 0, stream>>>(logits, red2, V);
    k_lsm_apply<<<(V + 255) / 256, 256, 0, stream>>>(logits, red2, outv, V);
}
